// MolGNN_6777458393679
// MI455X (gfx1250) — matrix operand order repaired and checked by reading, not compiled, not run on hardware
//
#include <hip/hip_runtime.h>

#define HID 128
#define OUTD 256
#define ZSTRIDE 132   // activation tile row stride (ushorts): breaks 64-bank wraparound
#define WSTRIDE 132   // weight tile row stride (ushorts): 256B row + 8B TDM pad = 264B (8B-aligned rows)

typedef __attribute__((ext_vector_type(16))) __bf16 v16bf;
typedef __attribute__((ext_vector_type(8)))  float  v8f;
typedef __attribute__((ext_vector_type(4)))  unsigned int v4u;
typedef __attribute__((ext_vector_type(8)))  int    v8i;
typedef __attribute__((ext_vector_type(4)))  int    v4i;
typedef unsigned short u16;
typedef unsigned int   u32;

static __device__ __forceinline__ __bf16 bits2bf(u16 u) {
    union { u16 u; __bf16 b; } c; c.u = u; return c.b;
}
static __device__ __forceinline__ u16 f2bf(float f) {  // round-to-nearest-even f32 -> bf16 bits
    union { float f; u32 u; } c; c.f = f;
    u32 r = c.u + 0x7FFFu + ((c.u >> 16) & 1u);
    return (u16)(r >> 16);
}
static __device__ __forceinline__ void atomicAddF(float* p, float v) {
    __hip_atomic_fetch_add(p, v, __ATOMIC_RELAXED, __HIP_MEMORY_SCOPE_AGENT);
}
// generic pointer to LDS: low 32 bits are the LDS byte offset (ISA 10.2 aperture rules)
static __device__ __forceinline__ u32 lds_offset(const void* p) {
    return (u32)(size_t)p;
}

// ---- TDM: DMA a 128x128 bf16 row-major tile (global) into LDS, +2 DWORD pad per 256B row ----
// D# per ISA 8.3/8.4. Group0: count=1 | lds_addr | global_addr | type=2.
// Group1: data_size=2B, pad_enable, pad_interval=64 DWORDs (code 5), pad_amount=2 DWORDs (code 1),
//         tensor_dim0=128, tensor_dim1=128, tile_dim0=128, tile_dim1=128, dim0_stride=128.
static __device__ __forceinline__ void tdm_load_w(const u16* gsrc, u32 ldsOff) {
    unsigned long long ga = (unsigned long long)(size_t)gsrc;
    v4u g0;
    g0.x = 1u;                                               // count=1 (valid user D#)
    g0.y = ldsOff;                                           // lds_addr (bytes)
    g0.z = (u32)ga;                                          // global_addr[31:0]
    g0.w = (u32)((ga >> 32) & 0x01FFFFFFull) | (2u << 30);   // global_addr[56:32] | type=2
    v8i g1;
    g1[0] = (int)((1u << 16) | (1u << 20) | (5u << 22) | (1u << 25)); // 2B | pad_en | 64dw interval | 2dw pad
    g1[1] = (int)(128u << 16);                               // tensor_dim0 = 128 (bits 79:48)
    g1[2] = (int)(128u << 16);                               // tensor_dim1 = 128 (bits 111:80)
    g1[3] = (int)(128u << 16);                               // tile_dim0  = 128 (bits 127:112)
    g1[4] = 128;                                             // tile_dim1 = 128, tile_dim2 = 0
    g1[5] = 128;                                             // tensor_dim0_stride = 128
    g1[6] = 0;                                               // tensor_dim1_stride (unused, 2D tile)
    g1[7] = 0;
    v4i gz4 = { 0, 0, 0, 0 };                                // groups 2/3 unused (<=2D)
    v8i gz8 = { 0, 0, 0, 0, 0, 0, 0, 0 };                    // 6-arg toolchain extra group
    __builtin_amdgcn_tensor_load_to_lds(g0, g1, gz4, gz4, gz8, 0);
}

// ---------------- embedding: h[n] = sum_i node_tables[i][x[n,i]] ----------------
__global__ __launch_bounds__(256) void node_embed_kernel(
    const int* __restrict__ x,
    const float* __restrict__ t0, const float* __restrict__ t1, const float* __restrict__ t2,
    const float* __restrict__ t3, const float* __restrict__ t4, const float* __restrict__ t5,
    const float* __restrict__ t6, const float* __restrict__ t7, const float* __restrict__ t8,
    float* __restrict__ h, int nN)
{
    long t = (long)blockIdx.x * blockDim.x + threadIdx.x;
    int n = (int)(t >> 5);
    if (n >= nN) return;
    int f = ((int)t & 31) * 4;          // one wave per node, 4 floats per lane (coalesced 512B)
    const int* xr = x + (long)n * 9;
    const float* tabs[9] = { t0,t1,t2,t3,t4,t5,t6,t7,t8 };
    float a0 = 0.f, a1 = 0.f, a2 = 0.f, a3 = 0.f;
#pragma unroll
    for (int i = 0; i < 9; ++i) {
        const float* row = tabs[i] + (long)xr[i] * HID + f;
        a0 += row[0]; a1 += row[1]; a2 += row[2]; a3 += row[3];
    }
    float* o = h + (long)n * HID + f;
    o[0] = a0; o[1] = a1; o[2] = a2; o[3] = a3;
}

// ---------------- z = h (copy, float4) ----------------
__global__ __launch_bounds__(256) void copy_kernel(const float4* __restrict__ s,
                                                   float4* __restrict__ d, long n4)
{
    long i = (long)blockIdx.x * blockDim.x + threadIdx.x;
    if (i < n4) d[i] = s[i];
}

// ---------------- messages: z[dst] += relu(h[src] + edge_emb) ----------------
__global__ __launch_bounds__(256) void msg_kernel(
    const int* __restrict__ eidx, const int* __restrict__ eattr,
    const float* __restrict__ e0, const float* __restrict__ e1, const float* __restrict__ e2,
    const float* __restrict__ h, float* __restrict__ z, int nE)
{
    long t = (long)blockIdx.x * blockDim.x + threadIdx.x;
    int e = (int)(t >> 5);
    if (e >= nE) return;
    int f = ((int)t & 31) * 4;          // one wave per edge
    int src = eidx[e];
    int dst = eidx[(long)nE + e];
    int a0 = eattr[(long)e * 3 + 0];
    int a1 = eattr[(long)e * 3 + 1];
    int a2 = eattr[(long)e * 3 + 2];
    const float* r0 = e0 + (long)a0 * HID + f;
    const float* r1 = e1 + (long)a1 * HID + f;
    const float* r2 = e2 + (long)a2 * HID + f;
    const float* hv = h + (long)src * HID + f;
    float* zp = z + (long)dst * HID + f;
#pragma unroll
    for (int c = 0; c < 4; ++c) {
        float m = hv[c] + r0[c] + r1[c] + r2[c];
        m = fmaxf(m, 0.f);
        atomicAddF(zp + c, m);
    }
}

// ---------------- weight f32 -> bf16, transposed (o[n*128+k] = w[k*128+n]) ----------------
__global__ __launch_bounds__(256) void wcvt_t_kernel(const float* __restrict__ w,
                                                     u16* __restrict__ o, int n)
{
    int i = blockIdx.x * blockDim.x + threadIdx.x;
    if (i < n) o[(i & 127) * HID + (i >> 7)] = f2bf(w[i]);
}

// ---------------- WMMA fragment builders (ISA 7.12.2 layouts) ----------------
// A 16x32 bf16: lane (half,r); elem pair j at K = (j<4 ? 2j : 8+2j) + 8*half
static __device__ __forceinline__ v16bf load_afrag(const u16* row, int ks, int half) {
    const u32* p = (const u32*)row;          // row stride is even -> u32 aligned
    v16bf a;
#pragma unroll
    for (int j = 0; j < 8; ++j) {
        int kb = (j < 4 ? 2 * j : 8 + 2 * j) + 8 * half + 32 * ks;
        u32 w = p[kb >> 1];
        a[2 * j]     = bits2bf((u16)w);
        a[2 * j + 1] = bits2bf((u16)(w >> 16));
    }
    return a;
}
// B 32x16 bf16 from K-contiguous (transposed) LDS: lane r = N; elem pair j at K = (j<4 ? 2j : 8+2j) + 8*half
static __device__ __forceinline__ v16bf load_bfrag(const u16* col, int ks, int half) {
    const u32* p = (const u32*)col;
    v16bf b;
#pragma unroll
    for (int j = 0; j < 8; ++j) {
        int kb = (j < 4 ? 2 * j : 8 + 2 * j) + 8 * half + 32 * ks;
        u32 w = p[kb >> 1];
        b[2 * j]     = bits2bf((u16)w);
        b[2 * j + 1] = bits2bf((u16)(w >> 16));
    }
    return b;
}

// ---------------- fused 2-layer MLP: h = relu(relu(z@W1+b1)@W2+b2) ----------------
// 128 threads = 4 waves; each wave owns a 16-row tile; K=128 -> 4 WMMA k-steps; 8 N-tiles.
// Weights (pre-transposed bf16 in global) are DMA'd into LDS by the Tensor Data Mover.
__global__ __launch_bounds__(128) void mlp_kernel(
    const float* __restrict__ z, float* __restrict__ h,
    const u16* __restrict__ w1t, const float* __restrict__ b1,
    const u16* __restrict__ w2t, const float* __restrict__ b2, int nrows)
{
    __shared__ u16 Wlds[HID * WSTRIDE];         // W^T tile: Wlds[n*WSTRIDE + k] (TDM pads rows)
    __shared__ u16 ybuf[4][16 * ZSTRIDE];       // per-wave activation tile (bf16)

    const int tid  = threadIdx.x;
    const int wave = tid >> 5;
    const int lane = tid & 31;
    const int half = lane >> 4;
    const int r    = lane & 15;
    const long rowBase = (long)blockIdx.x * 64 + (long)wave * 16;

    // wave 0 kicks the W1 DMA; meanwhile all waves stage their activation tiles
    if (tid < 32) tdm_load_w(w1t, lds_offset(Wlds));

    // stage this wave's 16 z rows as bf16 (coalesced f32 loads, converted in-flight)
    u16* zb = &ybuf[wave][0];
    for (int i = lane; i < 16 * HID; i += 32) {
        long row = rowBase + (i >> 7);
        if (row >= nrows) row = nrows - 1;      // clamp: WMMA needs full EXEC; stores are guarded
        zb[(i >> 7) * ZSTRIDE + (i & 127)] = f2bf(z[row * HID + (i & 127)]);
    }
    if (tid < 32) __builtin_amdgcn_s_wait_tensorcnt(0);
    __syncthreads();                            // publish W1 tile to all waves

    // GEMM1: y = z @ W1
    v8f acc[8] = {};
#pragma unroll
    for (int ks = 0; ks < 4; ++ks) {
        v16bf a = load_afrag(zb + r * ZSTRIDE, ks, half);
#pragma unroll
        for (int t8 = 0; t8 < 8; ++t8) {
            v16bf b = load_bfrag(Wlds + (16 * t8 + r) * WSTRIDE, ks, half);
            acc[t8] = __builtin_amdgcn_wmma_f32_16x16x32_bf16(
                false, a, false, b, (short)0, acc[t8], false, false);
        }
    }
    // epilogue 1: bias + relu, redistribute y into per-wave LDS (bf16) for GEMM2's A
#pragma unroll
    for (int t8 = 0; t8 < 8; ++t8) {
        float bias = b1[16 * t8 + r];
#pragma unroll
        for (int j = 0; j < 8; ++j) {
            float v = acc[t8][j] + bias;
            v = fmaxf(v, 0.f);
            zb[(j + 8 * half) * ZSTRIDE + 16 * t8 + r] = f2bf(v);  // D: M=j+8*half, N=r
        }
    }
    __syncthreads();                             // all waves done reading W1
    if (tid < 32) {                              // DMA W2 over W1
        tdm_load_w(w2t, lds_offset(Wlds));
        __builtin_amdgcn_s_wait_tensorcnt(0);
    }
    __syncthreads();                             // publish W2 tile

    // GEMM2: h = relu(y @ W2 + b2)
    v8f acc2[8] = {};
#pragma unroll
    for (int ks = 0; ks < 4; ++ks) {
        v16bf a = load_afrag(zb + r * ZSTRIDE, ks, half);
#pragma unroll
        for (int t8 = 0; t8 < 8; ++t8) {
            v16bf b = load_bfrag(Wlds + (16 * t8 + r) * WSTRIDE, ks, half);
            acc2[t8] = __builtin_amdgcn_wmma_f32_16x16x32_bf16(
                false, a, false, b, (short)0, acc2[t8], false, false);
        }
    }
#pragma unroll
    for (int t8 = 0; t8 < 8; ++t8) {
        float bias = b2[16 * t8 + r];
#pragma unroll
        for (int j = 0; j < 8; ++j) {
            float v = fmaxf(acc2[t8][j] + bias, 0.f);
            long row = rowBase + j + 8 * half;
            if (row < nrows) h[row * HID + 16 * t8 + r] = v;
        }
    }
}

// ---------------- pooling helpers ----------------
__global__ __launch_bounds__(256) void zero_kernel(float* __restrict__ p, long n)
{
    long i = (long)blockIdx.x * blockDim.x + threadIdx.x;
    if (i < n) p[i] = 0.f;
}

__global__ __launch_bounds__(256) void pool_kernel(const int* __restrict__ bidx,
    const float* __restrict__ h, float* __restrict__ g, int nN)
{
    long t = (long)blockIdx.x * blockDim.x + threadIdx.x;
    int n = (int)(t >> 5);
    if (n >= nN) return;
    int f = ((int)t & 31) * 4;
    int b = bidx[n];
    const float* hv = h + (long)n * HID + f;
    float* gp = g + (long)b * HID + f;
#pragma unroll
    for (int c = 0; c < 4; ++c) atomicAddF(gp + c, hv[c]);
}

// ---------------- projection + L2 normalize: one block per graph ----------------
__global__ __launch_bounds__(256) void proj_kernel(const float* __restrict__ g,
    const float* __restrict__ W, const float* __restrict__ b,
    float* __restrict__ out, int G)
{
    __shared__ float gs[HID];
    __shared__ float red[OUTD];
    int gi = blockIdx.x;
    int o = threadIdx.x;
    if (o < HID) gs[o] = g[(long)gi * HID + o];
    __syncthreads();
    float acc = b[o];
#pragma unroll 8
    for (int k = 0; k < HID; ++k) acc = fmaf(gs[k], W[(long)k * OUTD + o], acc);
    red[o] = acc * acc;
    __syncthreads();
    for (int s = 128; s > 0; s >>= 1) {
        if (o < s) red[o] += red[o + s];
        __syncthreads();
    }
    float denom = fmaxf(sqrtf(red[0]), 1e-12f);
    out[(long)gi * OUTD + o] = acc / denom;
}

extern "C" void kernel_launch(void* const* d_in, const int* in_sizes, int n_in,
                              void* d_out, int out_size, void* d_ws, size_t ws_size,
                              hipStream_t stream)
{
    const int* x     = (const int*)d_in[0];
    const int* eattr = (const int*)d_in[1];
    const int* eidx  = (const int*)d_in[2];
    const int* bidx  = (const int*)d_in[3];
    // d_in[4] = num_graphs scalar (derived from out_size instead)
    const float* nt[9];
    for (int i = 0; i < 9; ++i) nt[i] = (const float*)d_in[5 + i];
    const float* et[3];
    for (int i = 0; i < 3; ++i) et[i] = (const float*)d_in[14 + i];
    const float *W1[3], *B1[3], *W2[3], *B2[3];
    for (int l = 0; l < 3; ++l) {
        W1[l] = (const float*)d_in[17 + 4 * l];
        B1[l] = (const float*)d_in[18 + 4 * l];
        W2[l] = (const float*)d_in[19 + 4 * l];
        B2[l] = (const float*)d_in[20 + 4 * l];
    }
    const float* projW = (const float*)d_in[29];
    const float* projB = (const float*)d_in[30];

    const int nN = in_sizes[0] / 9;
    const int nE = in_sizes[1] / 3;
    const int G  = out_size / OUTD;
    float* out = (float*)d_out;

    // workspace layout: h | z | g | transposed bf16 weights
    char* ws = (char*)d_ws;
    size_t hB = (size_t)nN * HID * sizeof(float);
    float* h = (float*)ws;
    float* z = (float*)(ws + hB);
    float* g = (float*)(ws + 2 * hB);
    u16* wbf = (u16*)(ws + 2 * hB + (size_t)G * HID * sizeof(float));

    // convert + transpose weights once (W^T row-major = what the TDM tile copy wants)
    for (int l = 0; l < 3; ++l) {
        wcvt_t_kernel<<<64, 256, 0, stream>>>(W1[l], wbf + (size_t)l * 32768, HID * HID);
        wcvt_t_kernel<<<64, 256, 0, stream>>>(W2[l], wbf + (size_t)l * 32768 + 16384, HID * HID);
    }

    const long tN = (long)nN * 32;                   // one wave per node
    node_embed_kernel<<<(int)((tN + 255) / 256), 256, 0, stream>>>(
        x, nt[0], nt[1], nt[2], nt[3], nt[4], nt[5], nt[6], nt[7], nt[8], h, nN);

    const long n4 = (long)nN * (HID / 4);
    const long tE = (long)nE * 32;                   // one wave per edge
    for (int l = 0; l < 3; ++l) {
        copy_kernel<<<(int)((n4 + 255) / 256), 256, 0, stream>>>(
            (const float4*)h, (float4*)z, n4);       // z = h (GINE eps=0 self term)
        msg_kernel<<<(int)((tE + 255) / 256), 256, 0, stream>>>(
            eidx, eattr, et[0], et[1], et[2], h, z, nE);
        mlp_kernel<<<(nN + 63) / 64, 128, 0, stream>>>(
            z, h, wbf + (size_t)l * 32768, B1[l],
            wbf + (size_t)l * 32768 + 16384, B2[l], nN);
    }

    const long gElems = (long)G * HID;
    zero_kernel<<<(int)((gElems + 255) / 256), 256, 0, stream>>>(g, gElems);
    pool_kernel<<<(int)((tN + 255) / 256), 256, 0, stream>>>(bidx, h, g, nN);
    proj_kernel<<<G, 256, 0, stream>>>(g, projW, projB, out, G);
}
